// polyMultiHeadedRelAttention_72816875537088
// MI455X (gfx1250) — matrix operand order repaired and checked by reading, not compiled, not run on hardware
//
#include <hip/hip_runtime.h>
#include <hip/hip_bf16.h>

// Problem constants
#define BB    64
#define AA    128
#define DD    512
#define HH    8
#define EE    32768
#define LL    5
#define MAXDD 64
#define EDIMM 32
#define NN    (BB * AA)      // 8192
#define DHH   (DD / HH)      // 64

typedef __attribute__((ext_vector_type(16))) __bf16 v16bf;
typedef __attribute__((ext_vector_type(8)))  float  v8f;

// ---------------------------------------------------------------------------
// WMMA fragment loaders (16x16x32 bf16, wave32; A/B use this code's contiguous k order, not AMD's table order)
// A (16x32, MxK): lane&15 = row M, elems e -> K = (lane>>4)*16 + e
// B (32x16, KxN): lane&15 = col N, elems e -> K = (lane>>4)*16 + e
// C/D (16x16 f32): elem i -> M = i + 8*(lane>>4), N = lane&15
// ---------------------------------------------------------------------------

__device__ __forceinline__ v16bf load_A_bf16(const __bf16* X, int ld, int row0, int k0) {
    const int lane = threadIdx.x & 31;
    const __bf16* p = X + (size_t)(row0 + (lane & 15)) * ld + k0 + ((lane >> 4) << 4);
    v16bf a;
#pragma unroll
    for (int e = 0; e < 8; ++e) a[e] = p[e];
#pragma unroll
    for (int e = 0; e < 8; ++e) a[8 + e] = p[8 + e];
    return a;
}

// B fragment for X @ W^T:  Bfrag[k][n] = W[n][k]  (W row-major, ld = row stride)
__device__ __forceinline__ v16bf load_Bt_bf16(const __bf16* W, int ld, int n0, int k0) {
    const int lane = threadIdx.x & 31;
    const __bf16* p = W + (size_t)(n0 + (lane & 15)) * ld + k0 + ((lane >> 4) << 4);
    v16bf b;
#pragma unroll
    for (int e = 0; e < 16; ++e) b[e] = p[e];
    return b;
}

__device__ __forceinline__ v8f wmma_bf16(v16bf a, v16bf b, v8f c) {
    return __builtin_amdgcn_wmma_f32_16x16x32_bf16(
        /*neg_a=*/false, a, /*neg_b=*/false, b,
        /*c_mod=*/(short)0, c, /*reuse_a=*/false, /*reuse_b=*/false);
}

// ---------------------------------------------------------------------------
// Kernel 0: f32 -> bf16 conversion (done once; removes cvt work and halves
// bytes in every GEMM inner loop).  n4 = element count / 4.
// ---------------------------------------------------------------------------
__global__ __launch_bounds__(256) void cvt_bf16_kernel(
    const float* __restrict__ src, __bf16* __restrict__ dst, int n4)
{
    const int i = blockIdx.x * 256 + threadIdx.x;
    if (i >= n4) return;
    const float4 v = ((const float4*)src)[i];
    dst[4 * i + 0] = (__bf16)v.x;
    dst[4 * i + 1] = (__bf16)v.y;
    dst[4 * i + 2] = (__bf16)v.z;
    dst[4 * i + 3] = (__bf16)v.w;
}

// ---------------------------------------------------------------------------
// Kernel 1: fused QKV projection, bf16 in / bf16 out, 64x64 wave tile.
// Xbf(8192x512) @ W{q,k,v}^T + b.  q scaled by 1/sqrt(DH).
// q,k stored (B,H,A,DH); v stored transposed (B,H,DH,A) so the attention
// ctx GEMM gets contiguous B-fragments.
// waves: 3 * (8192/64) * (512/64) = 3072; 4 waves / block.
// ---------------------------------------------------------------------------
__global__ __launch_bounds__(128) void qkv_kernel(
    const __bf16* __restrict__ Xbf,
    const __bf16* __restrict__ Wqb, const float* __restrict__ bq,
    const __bf16* __restrict__ Wkb, const float* __restrict__ bk,
    const __bf16* __restrict__ Wvb, const float* __restrict__ bv,
    __bf16* __restrict__ qd, __bf16* __restrict__ kd, __bf16* __restrict__ vdT)
{
    const int wid = blockIdx.x * 4 + (threadIdx.x >> 5);
    const int wavesPerMat = (NN / 64) * (DD / 64);   // 128 * 8 = 1024
    const int m2 = wid / wavesPerMat;
    const int r  = wid - m2 * wavesPerMat;
    const int mg = r / (DD / 64);                    // 0..127  (rows mg*64..)
    const int ng = r - mg * (DD / 64);               // 0..7    (cols ng*64..)

    const __bf16* W; const float* bias; float scale; bool isV = false;
    if (m2 == 0)      { W = Wqb; bias = bq; scale = 0.125f; }
    else if (m2 == 1) { W = Wkb; bias = bk; scale = 1.0f;   }
    else              { W = Wvb; bias = bv; scale = 1.0f;   isV = true; }

    v8f acc[4][4];
#pragma unroll
    for (int i = 0; i < 4; ++i)
#pragma unroll
        for (int j = 0; j < 4; ++j)
            acc[i][j] = (v8f){0.f,0.f,0.f,0.f,0.f,0.f,0.f,0.f};

    for (int k0 = 0; k0 < DD; k0 += 32) {
        v16bf bf[4];
#pragma unroll
        for (int j = 0; j < 4; ++j)
            bf[j] = load_Bt_bf16(W, DD, ng * 64 + j * 16, k0);
#pragma unroll
        for (int i = 0; i < 4; ++i) {
            v16bf a = load_A_bf16(Xbf, DD, mg * 64 + i * 16, k0);
#pragma unroll
            for (int j = 0; j < 4; ++j)
                acc[i][j] = wmma_bf16(a, bf[j], acc[i][j]);
        }
    }

    const int lane = threadIdx.x & 31;
#pragma unroll
    for (int j = 0; j < 4; ++j) {
        const int n  = ng * 64 + j * 16 + (lane & 15);   // output channel
        const float bv_ = bias[n];
        const int h  = n >> 6;                           // DH = 64
        const int dh = n & 63;
#pragma unroll
        for (int i = 0; i < 4; ++i) {
            const int mbase = mg * 64 + i * 16 + ((lane >> 4) << 3);
#pragma unroll
            for (int e = 0; e < 8; ++e) {
                const int rrow = mbase + e;
                const int bI = rrow >> 7;                // A = 128
                const int aI = rrow & 127;
                const __bf16 val = (__bf16)((acc[i][j][e] + bv_) * scale);
                if (isV)
                    vdT[(((size_t)bI * HH + h) * DHH + dh) * AA + aI] = val;
                else if (m2 == 0)
                    qd[(((size_t)bI * HH + h) * AA + aI) * DHH + dh] = val;
                else
                    kd[(((size_t)bI * HH + h) * AA + aI) * DHH + dh] = val;
            }
        }
    }
}

// ---------------------------------------------------------------------------
// Kernel 2a: edge step scores  esc[e][l] = sum_k edge_attr[e][k] * w_edge[l][k]
// ---------------------------------------------------------------------------
__global__ __launch_bounds__(256) void edge_score_kernel(
    const float* __restrict__ edge_attr, const float* __restrict__ w_edge,
    float* __restrict__ esc)
{
    const int e = blockIdx.x * 256 + threadIdx.x;
    if (e >= EE) return;
    float s[LL] = {0.f, 0.f, 0.f, 0.f, 0.f};
    const float* row = edge_attr + (size_t)e * EDIMM;
#pragma unroll
    for (int k = 0; k < EDIMM; ++k) {
        const float x = row[k];
#pragma unroll
        for (int l = 0; l < LL; ++l) s[l] += x * w_edge[l * EDIMM + k];
    }
#pragma unroll
    for (int l = 0; l < LL; ++l) esc[(size_t)e * LL + l] = s[l];
}

// ---------------------------------------------------------------------------
// Kernel 2b: node step scores  nsc[n][l] = sum_k poly_vec[n][k] * w_node[l][k]
// ---------------------------------------------------------------------------
__global__ __launch_bounds__(256) void node_score_kernel(
    const float* __restrict__ poly_vec, const float* __restrict__ w_node,
    float* __restrict__ nsc)
{
    const int n = blockIdx.x * 256 + threadIdx.x;
    if (n >= NN) return;
    float s[LL] = {0.f, 0.f, 0.f, 0.f, 0.f};
    const float* row = poly_vec + (size_t)n * DD;
    for (int k = 0; k < DD; ++k) {
        const float x = row[k];
#pragma unroll
        for (int l = 0; l < LL; ++l) s[l] += x * w_node[l * DD + k];
    }
#pragma unroll
    for (int l = 0; l < LL; ++l) nsc[(size_t)n * LL + l] = s[l];
}

// ---------------------------------------------------------------------------
// Kernel 3a: pe channels 1..3 + zero channel 0.  pe layout: (B, 4, A, A) f32
// ---------------------------------------------------------------------------
__global__ __launch_bounds__(256) void pe_kernel(
    const int* __restrict__ distances, const float* __restrict__ dist_table,
    const int* __restrict__ edge_paths, const float* __restrict__ esc,
    const int* __restrict__ node_paths, const float* __restrict__ nsc,
    float* __restrict__ pe)
{
    const int idx = blockIdx.x * 256 + threadIdx.x;
    if (idx >= BB * AA * AA) return;
    const int b  = idx / (AA * AA);
    const int ij = idx - b * (AA * AA);
    const size_t base = ((size_t)b * 4) * AA * AA + ij;

    pe[base] = 0.0f;                              // adjacency, scattered later

    int d = distances[idx];
    d = d < 0 ? 0 : (d > MAXDD - 1 ? MAXDD - 1 : d);
    pe[base + (size_t)1 * AA * AA] = dist_table[d];

    float num = 0.f; int cnt = 0;
#pragma unroll
    for (int l = 0; l < LL; ++l) {
        const int p = edge_paths[(size_t)idx * LL + l];
        if (p >= 0) { num += esc[(size_t)p * LL + l]; ++cnt; }
    }
    pe[base + (size_t)2 * AA * AA] = num / (float)(cnt > 0 ? cnt : 1);

    num = 0.f; cnt = 0;
#pragma unroll
    for (int l = 0; l < LL; ++l) {
        const int p = node_paths[(size_t)idx * LL + l];
        if (p >= 0) { num += nsc[(size_t)p * LL + l]; ++cnt; }
    }
    pe[base + (size_t)3 * AA * AA] = num / (float)(cnt > 0 ? cnt : 1);
}

// ---------------------------------------------------------------------------
// Kernel 3b: adjacency scatter into pe channel 0.
// ---------------------------------------------------------------------------
__global__ __launch_bounds__(256) void adj_kernel(
    const int* __restrict__ edge_index, const int* __restrict__ ptr,
    const int* __restrict__ batch, float* __restrict__ pe)
{
    const int e = blockIdx.x * 256 + threadIdx.x;
    if (e >= EE) return;
    const int s  = edge_index[e];
    const int dn = edge_index[EE + e];
    const int g  = batch[s];
    const int sl = s - ptr[g];
    const int dl = dn - ptr[g];
    atomicAdd(&pe[(((size_t)g * 4) * AA + sl) * AA + dl], 1.0f);
}

// ---------------------------------------------------------------------------
// Kernel 4: attention.  One block per (b,h); 8 waves; wave qt owns rows
// [qt*16, qt*16+16).  WMMA scores (Q frags hoisted), register softmax with
// shfl_xor within the 16-lane row groups, attn -> d_out, masked probs staged
// in LDS (bf16), WMMA ctx against V^T (contiguous B frags), ctx out as bf16.
// ---------------------------------------------------------------------------
__global__ __launch_bounds__(256) void attn_kernel(
    const __bf16* __restrict__ qbf, const __bf16* __restrict__ kbf,
    const __bf16* __restrict__ vbfT, const float* __restrict__ pe,
    const unsigned char* __restrict__ mask,
    float* __restrict__ attn_out, __bf16* __restrict__ ctx)
{
    __shared__ __bf16 lds_p[8][16 * AA];

    const int bh   = blockIdx.x;          // 0 .. B*H-1
    const int b    = bh / HH;
    const int h    = bh - b * HH;
    const int wave = threadIdx.x >> 5;
    const int lane = threadIdx.x & 31;
    const int m0   = wave * 16;

    const __bf16* qb = qbf  + (size_t)bh * AA * DHH;
    const __bf16* kb = kbf  + (size_t)bh * AA * DHH;
    const __bf16* vb = vbfT + (size_t)bh * DHH * AA;    // (DH, A)
    const float*  peb = pe + ((size_t)b * 4 + (h >> 1)) * AA * AA;
    const unsigned char* mk = mask + (size_t)b * AA * AA;

    const int ncol  = lane & 15;
    const int mbase = m0 + ((lane >> 4) << 3);

    // ---- scores: Q fragments are loop-invariant -> load once ----
    const v16bf aq0 = load_A_bf16(qb, DHH, m0, 0);
    const v16bf aq1 = load_A_bf16(qb, DHH, m0, 32);

    v8f sc[8];
#pragma unroll
    for (int nt = 0; nt < 8; ++nt) {
        v8f acc = {0.f,0.f,0.f,0.f,0.f,0.f,0.f,0.f};
        acc = wmma_bf16(aq0, load_Bt_bf16(kb, DHH, nt * 16, 0),  acc);
        acc = wmma_bf16(aq1, load_Bt_bf16(kb, DHH, nt * 16, 32), acc);
        const int n = nt * 16 + ncol;
#pragma unroll
        for (int i = 0; i < 8; ++i) {
            const int m = mbase + i;
            float s = acc[i] + peb[m * AA + n];
            if (mk[m * AA + n]) s = -1e18f;
            acc[i] = s;
        }
        sc[nt] = acc;
    }

    // ---- row softmax (row m = mbase+i lives across 16 lanes of one half) ----
    float inv[8];
#pragma unroll
    for (int i = 0; i < 8; ++i) {
        float mx = -3.4e38f;
#pragma unroll
        for (int nt = 0; nt < 8; ++nt) mx = fmaxf(mx, sc[nt][i]);
#pragma unroll
        for (int off = 1; off < 16; off <<= 1)
            mx = fmaxf(mx, __shfl_xor(mx, off, 32));
        float sum = 0.f;
#pragma unroll
        for (int nt = 0; nt < 8; ++nt) {
            const float e = __expf(sc[nt][i] - mx);
            sc[nt][i] = e;
            sum += e;
        }
#pragma unroll
        for (int off = 1; off < 16; off <<= 1)
            sum += __shfl_xor(sum, off, 32);
        inv[i] = 1.0f / sum;
    }

    // ---- write attn (d_out) and stage masked probs in LDS as bf16 ----
    float* attn_bh = attn_out + (size_t)bh * AA * AA;
#pragma unroll
    for (int nt = 0; nt < 8; ++nt) {
        const int n = nt * 16 + ncol;
#pragma unroll
        for (int i = 0; i < 8; ++i) {
            const int m = mbase + i;
            const float p = sc[nt][i] * inv[i];
            attn_bh[m * AA + n] = p;
            const float pc = mk[m * AA + n] ? 0.0f : p;
            lds_p[wave][(m - m0) * AA + n] = (__bf16)pc;
        }
    }
    __syncthreads();

    // ---- ctx = p @ V : P fragments loop-invariant over nt -> load once ----
    v16bf ap[4];
#pragma unroll
    for (int t = 0; t < 4; ++t)
        ap[t] = load_A_bf16(&lds_p[wave][0], AA, 0, t * 32);

#pragma unroll
    for (int nt = 0; nt < 4; ++nt) {
        v8f acc = {0.f,0.f,0.f,0.f,0.f,0.f,0.f,0.f};
#pragma unroll
        for (int t = 0; t < 4; ++t)
            acc = wmma_bf16(ap[t], load_Bt_bf16(vb, AA, nt * 16, t * 32), acc);
        const int n = nt * 16 + ncol;
#pragma unroll
        for (int i = 0; i < 8; ++i) {
            const int m = mbase + i;
            ctx[((size_t)b * AA + m) * DD + h * DHH + n] = (__bf16)acc[i];
        }
    }
}

// ---------------------------------------------------------------------------
// Kernel 5: out = ctx_bf16(8192x512) @ Wo^T + bo -> d_out[0 : B*A*D] (f32).
// 64x64 wave tile, same blocking as qkv.
// ---------------------------------------------------------------------------
__global__ __launch_bounds__(128) void outproj_kernel(
    const __bf16* __restrict__ ctx, const __bf16* __restrict__ Wob,
    const float* __restrict__ bo, float* __restrict__ out)
{
    const int wid = blockIdx.x * 4 + (threadIdx.x >> 5);
    const int mg  = wid / (DD / 64);
    const int ng  = wid - mg * (DD / 64);

    v8f acc[4][4];
#pragma unroll
    for (int i = 0; i < 4; ++i)
#pragma unroll
        for (int j = 0; j < 4; ++j)
            acc[i][j] = (v8f){0.f,0.f,0.f,0.f,0.f,0.f,0.f,0.f};

    for (int k0 = 0; k0 < DD; k0 += 32) {
        v16bf bf[4];
#pragma unroll
        for (int j = 0; j < 4; ++j)
            bf[j] = load_Bt_bf16(Wob, DD, ng * 64 + j * 16, k0);
#pragma unroll
        for (int i = 0; i < 4; ++i) {
            v16bf a = load_A_bf16(ctx, DD, mg * 64 + i * 16, k0);
#pragma unroll
            for (int j = 0; j < 4; ++j)
                acc[i][j] = wmma_bf16(a, bf[j], acc[i][j]);
        }
    }

    const int lane = threadIdx.x & 31;
#pragma unroll
    for (int j = 0; j < 4; ++j) {
        const int n = ng * 64 + j * 16 + (lane & 15);
        const float bv = bo[n];
#pragma unroll
        for (int i = 0; i < 4; ++i) {
            const int mbase = mg * 64 + i * 16 + ((lane >> 4) << 3);
#pragma unroll
            for (int e = 0; e < 8; ++e)
                out[(size_t)(mbase + e) * DD + n] = acc[i][j][e] + bv;
        }
    }
}

// ---------------------------------------------------------------------------
// Host launcher
// ---------------------------------------------------------------------------
extern "C" void kernel_launch(void* const* d_in, const int* in_sizes, int n_in,
                              void* d_out, int out_size, void* d_ws, size_t ws_size,
                              hipStream_t stream)
{
    const float* poly_vec   = (const float*)d_in[0];
    const float* inputs     = (const float*)d_in[1];
    const unsigned char* mask = (const unsigned char*)d_in[2];
    const int*   edge_index = (const int*)d_in[3];
    const float* edge_attr  = (const float*)d_in[4];
    const int*   ptr        = (const int*)d_in[5];
    const int*   batch      = (const int*)d_in[6];
    const int*   distances  = (const int*)d_in[7];
    const int*   node_paths = (const int*)d_in[8];
    const int*   edge_paths = (const int*)d_in[9];
    const float* Wq = (const float*)d_in[10]; const float* bq = (const float*)d_in[11];
    const float* Wk = (const float*)d_in[12]; const float* bk = (const float*)d_in[13];
    const float* Wv = (const float*)d_in[14]; const float* bv = (const float*)d_in[15];
    const float* Wo = (const float*)d_in[16]; const float* bo = (const float*)d_in[17];
    const float* dist_table = (const float*)d_in[18];
    const float* w_edge     = (const float*)d_in[19];
    const float* w_node     = (const float*)d_in[20];

    float* out      = (float*)d_out;                        // (B, A, D)
    float* attn_out = (float*)d_out + (size_t)BB * AA * DD; // (B, H, A, A)

    // workspace carve-up
    char* w = (char*)d_ws;
    __bf16* qbf  = (__bf16*)w; w += (size_t)NN * DD * sizeof(__bf16);
    __bf16* kbf  = (__bf16*)w; w += (size_t)NN * DD * sizeof(__bf16);
    __bf16* vbfT = (__bf16*)w; w += (size_t)NN * DD * sizeof(__bf16);
    __bf16* ctxb = (__bf16*)w; w += (size_t)NN * DD * sizeof(__bf16);
    __bf16* xbf  = (__bf16*)w; w += (size_t)NN * DD * sizeof(__bf16);
    __bf16* wqb  = (__bf16*)w; w += (size_t)DD * DD * sizeof(__bf16);
    __bf16* wkb  = (__bf16*)w; w += (size_t)DD * DD * sizeof(__bf16);
    __bf16* wvb  = (__bf16*)w; w += (size_t)DD * DD * sizeof(__bf16);
    __bf16* wob  = (__bf16*)w; w += (size_t)DD * DD * sizeof(__bf16);
    float* pe    = (float*)w;  w += (size_t)BB * 4 * AA * AA * sizeof(float);
    float* esc   = (float*)w;  w += (size_t)EE * LL * sizeof(float);
    float* nsc   = (float*)w;  w += (size_t)NN * LL * sizeof(float);
    (void)ws_size; (void)in_sizes; (void)n_in; (void)out_size;

    // 0) one-time f32 -> bf16 conversions (activations + all weights)
    cvt_bf16_kernel<<<(NN * DD / 4 + 255) / 256, 256, 0, stream>>>(inputs, xbf, NN * DD / 4);
    cvt_bf16_kernel<<<(DD * DD / 4 + 255) / 256, 256, 0, stream>>>(Wq, wqb, DD * DD / 4);
    cvt_bf16_kernel<<<(DD * DD / 4 + 255) / 256, 256, 0, stream>>>(Wk, wkb, DD * DD / 4);
    cvt_bf16_kernel<<<(DD * DD / 4 + 255) / 256, 256, 0, stream>>>(Wv, wvb, DD * DD / 4);
    cvt_bf16_kernel<<<(DD * DD / 4 + 255) / 256, 256, 0, stream>>>(Wo, wob, DD * DD / 4);

    // 1) QKV projections: 3 * 1024 waves / 4 waves per block
    qkv_kernel<<<(3 * (NN / 64) * (DD / 64)) / 4, 128, 0, stream>>>(
        xbf, wqb, bq, wkb, bk, wvb, bv, qbf, kbf, vbfT);

    // 2) path step scores
    edge_score_kernel<<<EE / 256, 256, 0, stream>>>(edge_attr, w_edge, esc);
    node_score_kernel<<<NN / 256, 256, 0, stream>>>(poly_vec, w_node, nsc);

    // 3) positional-encoding channels + adjacency scatter
    pe_kernel<<<(BB * AA * AA) / 256, 256, 0, stream>>>(
        distances, dist_table, edge_paths, esc, node_paths, nsc, pe);
    adj_kernel<<<EE / 256, 256, 0, stream>>>(edge_index, ptr, batch, pe);

    // 4) attention (scores + softmax + attn output + bf16 ctx)
    attn_kernel<<<BB * HH, 256, 0, stream>>>(qbf, kbf, vbfT, pe, mask, attn_out, ctxb);

    // 5) output projection
    outproj_kernel<<<((NN / 64) * (DD / 64)) / 4, 128, 0, stream>>>(ctxb, wob, bo, out);
}
